// SelectionLstm_38577396252947
// MI455X (gfx1250) — matrix operand order repaired and checked by reading, not compiled, not run on hardware
//
#include <hip/hip_runtime.h>
#include <hip/hip_bf16.h>
#include <stdint.h>

// ---------------------------------------------------------------- types
typedef __attribute__((ext_vector_type(16))) _Float16 v16h;
typedef __attribute__((ext_vector_type(8)))  float    v8f;
typedef __attribute__((ext_vector_type(4)))  _Float16 v4h;
typedef __attribute__((ext_vector_type(4)))  unsigned int v4u;
typedef __attribute__((ext_vector_type(8)))  int      v8i;
typedef __attribute__((ext_vector_type(4)))  int      v4i;

#define TT    2000
#define BB    64
#define HH    512
#define NBLK  16      // persistent workgroups (== max cluster size)
#define NTHR  512     // 16 waves of 32

__device__ __forceinline__ float hc_gate(float a) {
  // hard-concrete eval gate: clamp(sigmoid(a) * (zeta-gamma) + gamma, 0, 1)
  float s = 1.0f / (1.0f + __expf(-a));
  s = s * (1.1f - (-0.1f)) + (-0.1f);
  return fminf(fmaxf(s, 0.0f), 1.0f);
}

// Branch-free activations: v_exp_f32 + v_rcp_f32 (TRANS ops co-execute with WMMA)
__device__ __forceinline__ float sigmf(float x) {
  return __builtin_amdgcn_rcpf(1.0f + __expf(-x));
}
__device__ __forceinline__ float tanh_fast(float x) {
  // tanh(x) = 1 - 2/(e^{2x}+1); saturates correctly via exp overflow/underflow
  return 1.0f - 2.0f * __builtin_amdgcn_rcpf(1.0f + __expf(2.0f * x));
}

__device__ __forceinline__ v8f wmma_f16(v16h a, v16h b, v8f c) {
  return __builtin_amdgcn_wmma_f32_16x16x32_f16(false, a, false, b, (short)0, c, false, false);
}

union AFrag { uint4 u[2]; v16h h; };

// ------------------------------------------------ TDM: 2D f16 tile -> LDS
// D# per CDNA5 ISA ch.8: group0 (4 dw), group1 (8 dw), groups 2/3 zero (2D tensor).
// Loads a 512 x 64 (dim0 x dim1) f16 tile, stride0 = 512 elems, contiguous.
// 6-arg toolchain form: (v4u g0, v8i g1, v4i g2, v4i g3, v8i g4, i32 cpol)
__device__ __forceinline__ void tdm_load_64x512_f16(unsigned lds_off,
                                                    unsigned long long gaddr) {
  v4u g0;
  g0.x = 1u;                                             // count=1, user mode
  g0.y = lds_off;                                        // lds_addr (bytes)
  g0.z = (unsigned)(gaddr & 0xFFFFFFFFu);                // global_addr[31:0]
  g0.w = (unsigned)((gaddr >> 32) & 0x1FFFFFFu)          // global_addr[56:32]
       | (2u << 30);                                     // type = 2 ("image")
  v8i g1;
  g1[0] = (1 << 16);          // workgroup_mask=0, data_size=1 (2 bytes)
  g1[1] = (HH & 0xFFFF) << 16;       // tensor_dim0[15:0] = 512
  g1[2] = (BB & 0xFFFF) << 16;       // tensor_dim0 hi=0 | tensor_dim1=64
  g1[3] = (HH & 0xFFFF) << 16;       // tensor_dim1 hi=0 | tile_dim0=512
  g1[4] = BB;                        // tile_dim1=64, tile_dim2=0
  g1[5] = HH;                        // tensor_dim0_stride[31:0] = 512
  g1[6] = 0;                         // stride0 hi | stride1 lo
  g1[7] = 0;
  v4i z4 = {0, 0, 0, 0};
  v8i z8 = {0, 0, 0, 0, 0, 0, 0, 0};
  __builtin_amdgcn_tensor_load_to_lds(g0, g1, z4, z4, z8, 0);
}

// ---------------------------------------------------------- prep kernels
__global__ void zero_ws_kernel(unsigned int* p, int n) {
  int i = blockIdx.x * 256 + threadIdx.x;
  if (i < n) p[i] = 0u;
}

// Pack combined, scale-folded, column-permuted weights into WMMA B-fragment layout.
// Permuted output col n: gate = n&3, j = n>>2  -> orig row r = gate*512 + j.
// B[k][n]: k<512 -> W[r][k]*s[k];  k>=512 -> U[r][k-512]*s[j]*s[k-512].
// Fragment layout (this code's contiguous-k B order, A loads match; wave32):
//   frag f = nTile*32 + kb ; lane L<16: N=nTile*16+L, 16 halves = K kb*32+0..15
//                            lane L>=16: N=nTile*16+L-16, K kb*32+16..31
__global__ void pack_weights_kernel(const float* __restrict__ W,
                                    const float* __restrict__ U,
                                    const float* __restrict__ la,
                                    _Float16* __restrict__ pB) {
  int e = blockIdx.x * 256 + threadIdx.x;          // 0 .. 2048*1024-1
  if (e >= 2048 * 1024) return;
  int frag   = e >> 9;
  int within = e & 511;
  int lane   = within >> 4;
  int half   = within & 15;
  int nTile  = frag >> 5;
  int kb     = frag & 31;
  int N = nTile * 16 + (lane & 15);
  int K = kb * 32 + ((lane >= 16) ? 16 : 0) + half;
  int g = N & 3, j = N >> 2;
  int r = g * HH + j;
  float val;
  if (K < HH) {
    val = W[r * HH + K] * hc_gate(la[K]);
  } else {
    int k2 = K - HH;
    val = U[r * HH + k2] * hc_gate(la[j]) * hc_gate(la[k2]);
  }
  pB[e] = (_Float16)val;
}

__global__ void convert_x_kernel(const float* __restrict__ x,
                                 _Float16* __restrict__ x16, long long n) {
  long long i = ((long long)blockIdx.x * 256 + threadIdx.x) * 4;
  if (i + 3 >= n) return;
  float4 v = *(const float4*)(x + i);
  v4h o = {(_Float16)v.x, (_Float16)v.y, (_Float16)v.z, (_Float16)v.w};
  *(v4h*)(x16 + i) = o;
}

// ------------------------------------------------------- grid-wide sync
__device__ __forceinline__ void grid_sync(int* arrive, int* gen, int step) {
  __syncthreads();
  if (threadIdx.x == 0) {
    __threadfence();
    int ticket = __hip_atomic_fetch_add(arrive, 1, __ATOMIC_ACQ_REL,
                                        __HIP_MEMORY_SCOPE_AGENT);
    if (ticket == NBLK * (step + 1) - 1) {
      __hip_atomic_store(gen, step + 1, __ATOMIC_RELEASE, __HIP_MEMORY_SCOPE_AGENT);
    } else {
      while (__hip_atomic_load(gen, __ATOMIC_ACQUIRE, __HIP_MEMORY_SCOPE_AGENT)
             < step + 1) {
        __builtin_amdgcn_s_sleep(2);
      }
    }
  }
  // cluster barrier: NOP when not cluster-dispatched, real HW sync when it is
  __builtin_amdgcn_s_cluster_barrier();
  __syncthreads();
}

// -------------------------------------------------- persistent LSTM core
__global__ void __launch_bounds__(NTHR, 1)
lstm_persistent_kernel(const _Float16* __restrict__ x16,
                       const _Float16* __restrict__ pB,
                       const float* __restrict__ bvec,
                       float* __restrict__ out,
                       _Float16* __restrict__ hbuf,   // [2][64*512] f16
                       int* __restrict__ sync) {
  __shared__ __align__(16) _Float16 sX[BB * HH];   // 64 KB; reused as f32 gates
  __shared__ __align__(16) _Float16 sH[BB * HH];   // 64 KB
  __shared__ __align__(16) float sBias[128];

  const int tid  = threadIdx.x;
  const int wg   = blockIdx.x;
  const int lane = tid & 31;
  const int wid  = tid >> 5;                 // 0..15

  if (tid < 128) {                           // permuted bias slice for this WG
    int g = tid & 3, j = wg * 32 + (tid >> 2);
    sBias[tid] = bvec[g * HH + j];
  }

  // tile assignment: wave -> (nTileLocal = wid&7, mPair = wid>>3 -> mTiles {2p,2p+1})
  const int nTileG  = wg * 8 + (wid & 7);
  const int mPair   = wid >> 3;
  const int row0    = mPair * 32 + (lane & 15);
  const int halfSel = (lane >> 4) * 16;      // K sub-block select (A fragment)

  // consumer mapping: thread owns (batch bb, 4 consecutive hidden units)
  const int bb  = tid >> 3;
  const int jl0 = (tid & 7) * 4;
  float c0 = 0.f, c1 = 0.f, c2 = 0.f, c3 = 0.f;

  int phase = 0;
  for (int t = 0; t < TT; ++t) {
    // ---- async tensor loads: x_t and h into LDS (wave 0 drives the TDM)
    if (wid == 0) {
      tdm_load_64x512_f16((unsigned)(uintptr_t)(void*)sX,
                          (unsigned long long)(uintptr_t)(x16 + (size_t)t * BB * HH));
      tdm_load_64x512_f16((unsigned)(uintptr_t)(void*)sH,
                          (unsigned long long)(uintptr_t)(hbuf + (size_t)phase * BB * HH));
      __builtin_amdgcn_s_wait_tensorcnt(0);
    }
    __syncthreads();

    // ---- gates = [x_t | h] @ Wc^T  via v_wmma_f32_16x16x32_f16
    v8f acc0 = {}; v8f acc1 = {};
    #pragma unroll 4
    for (int kb = 0; kb < 32; ++kb) {
      const _Float16* asrc = (kb < 16) ? sX : sH;
      const int kloc = (kb * 32 + halfSel) & (HH - 1);
      AFrag a0, a1, bf;
      a0.u[0] = *(const uint4*)(asrc + row0 * HH + kloc);
      a0.u[1] = *(const uint4*)(asrc + row0 * HH + kloc + 8);
      a1.u[0] = *(const uint4*)(asrc + (row0 + 16) * HH + kloc);
      a1.u[1] = *(const uint4*)(asrc + (row0 + 16) * HH + kloc + 8);
      const uint4* bp = (const uint4*)(pB + (((size_t)(nTileG * 32 + kb)) << 9)
                                          + lane * 16);
      bf.u[0] = bp[0];
      bf.u[1] = bp[1];
      acc0 = wmma_f16(a0.h, bf.h, acc0);
      acc1 = wmma_f16(a1.h, bf.h, acc1);
    }
    __syncthreads();                         // everyone done reading sX/sH

    // ---- spill D fragments into LDS gate buffer (f32, reusing sX region)
    float* gatesF = (float*)sX;
    {
      const int rbase = mPair * 32 + ((lane < 16) ? 0 : 8);
      const int col   = (wid & 7) * 16 + (lane & 15);
      #pragma unroll
      for (int v = 0; v < 8; ++v) {
        gatesF[(rbase + v) * 128 + col]      = acc0[v];
        gatesF[(rbase + 16 + v) * 128 + col] = acc1[v];
      }
    }
    __syncthreads();

    // ---- elementwise LSTM cell update; c lives in registers across all t
    {
      float hq[4];
      #pragma unroll
      for (int e = 0; e < 4; ++e) {
        const int jl = jl0 + e;
        float4 gv = ((const float4*)gatesF)[bb * 32 + jl];   // i,f,g,o
        float4 bi = ((const float4*)sBias)[jl];
        float ig = sigmf(gv.x + bi.x);
        float fg = sigmf(gv.y + bi.y);
        float gg = tanh_fast(gv.z + bi.z);
        float og = sigmf(gv.w + bi.w);
        float cc = (e == 0) ? c0 : (e == 1) ? c1 : (e == 2) ? c2 : c3;
        cc = fg * cc + ig * gg;
        if (e == 0) c0 = cc; else if (e == 1) c1 = cc;
        else if (e == 2) c2 = cc; else c3 = cc;
        hq[e] = og * tanh_fast(cc);
      }
      const size_t obase = (size_t)t * (BB * HH) + (size_t)bb * HH + wg * 32 + jl0;
      float4 ho; ho.x = hq[0]; ho.y = hq[1]; ho.z = hq[2]; ho.w = hq[3];
      *(float4*)(out + obase) = ho;
      v4h h16 = {(_Float16)hq[0], (_Float16)hq[1], (_Float16)hq[2], (_Float16)hq[3]};
      *(v4h*)(hbuf + (size_t)(phase ^ 1) * BB * HH + (size_t)bb * HH + wg * 32 + jl0) = h16;
    }

    if (t + 1 < TT) grid_sync(sync, sync + 1, t);
    phase ^= 1;
  }
}

// ---------------------------------------------------------------- launch
extern "C" void kernel_launch(void* const* d_in, const int* in_sizes, int n_in,
                              void* d_out, int out_size, void* d_ws, size_t ws_size,
                              hipStream_t stream) {
  const float* x    = (const float*)d_in[0];   // [T,B,H]
  const float* W    = (const float*)d_in[1];   // [4H,H]
  const float* U    = (const float*)d_in[2];   // [4H,H]
  const float* bvec = (const float*)d_in[3];   // [4H]
  const float* la   = (const float*)d_in[4];   // [H]
  float* out = (float*)d_out;

  char* ws = (char*)d_ws;
  int*       syncp = (int*)ws;                                  // 8 B (zeroed)
  _Float16*  hbuf  = (_Float16*)(ws + 256);                     // 2*64*512*2 = 128 KB (zeroed)
  _Float16*  pB    = (_Float16*)(ws + 131584);                  // 4 MB packed weights
  _Float16*  x16   = (_Float16*)(ws + 131584 + 4194304);        // 125 MB x in f16

  // zero sync counters + both h phase buffers (bytes [0, 131328) -> 32832 dwords)
  zero_ws_kernel<<<129, 256, 0, stream>>>((unsigned int*)ws, 32832);
  pack_weights_kernel<<<(2048 * 1024) / 256, 256, 0, stream>>>(W, U, la, pB);
  convert_x_kernel<<<(TT * BB * HH) / (256 * 4), 256, 0, stream>>>(
      x, x16, (long long)TT * BB * HH);
  lstm_persistent_kernel<<<NBLK, NTHR, 0, stream>>>(x16, pB, bvec, out, hbuf, syncp);
}
